// BiMatchLoss_1623497637922
// MI455X (gfx1250) — matrix operand order repaired and checked by reading, not compiled, not run on hardware
//
#include <hip/hip_runtime.h>
#include <hip/hip_bf16.h>
#include <math.h>

typedef __attribute__((ext_vector_type(16))) _Float16 v16h;
typedef __attribute__((ext_vector_type(8)))  float    v8f;

#define B_  32
#define S_  1024
#define E_  6
#define C_  16
#define K_  (S_ * C_ * 2)        // 32768 reduction length for cost
#define TOKSTRIDE (E_ * C_ * 2)  // 192 floats per (b,s) token
#define KSPLIT 8                 // K-parallel blocks per batch
#define WAVES  8                 // waves per block (256 threads, wave32)
#define TOK_PER_BLK 64
#define BCE_THREADS 192          // 6 waves; one thread per intra-token offset

// ---------------------------------------------------------------------------
// Kernel 1: cost[b,t,o] = -sum_k tgt[b,t,k]*out[b,o,k] via V_WMMA_F32_16X16X32_F16
// Pad 6x6 -> 16x16. Pad rows/cols need NOT be zeroed: D[m<6][n<6] only reads
// A rows m<6 / B cols n<6, so pad lanes just re-load row 5 (clamped, in-bounds).
// k = s*32 + (c*2+i): each 32-wide k chunk is one contiguous 32-float run.
// ---------------------------------------------------------------------------
__global__ __launch_bounds__(256) void cost_wmma_kernel(
    const float* __restrict__ outputs, const float* __restrict__ targets,
    float* __restrict__ costp /* [B][KSPLIT][36] */) {
  __shared__ float partial[WAVES][256];

  const int b    = blockIdx.x;
  const int kb   = blockIdx.y;
  const int tid  = threadIdx.x;
  const int w    = tid >> 5;
  const int lane = tid & 31;
  const int mn   = lane & 15;             // A-row (t) / B-col (o) for this lane
  const int half = lane >> 4;
  const int mnc  = (mn < E_) ? mn : (E_ - 1);  // clamp pad lanes in-bounds

  const int kChunk = K_ / (KSPLIT * WAVES);    // 512 -> 16 WMMA steps of K=32
  const int k0base = (kb * WAVES + w) * kChunk;

  const float* pA = targets + (size_t)b * (S_ * TOKSTRIDE)
                           + (size_t)(k0base >> 5) * TOKSTRIDE + mnc * 32;
  const float* pB = outputs + (size_t)b * (S_ * TOKSTRIDE)
                           + (size_t)(k0base >> 5) * TOKSTRIDE + mnc * 32;
  const int aoff = 8 * half;    // A: K groups {8h+0..7, 16+8h+0..7}
  const int boff = 8 * half;    // B: K groups {8h+0..7, 16+8h+0..7}

  v8f acc = {};
#pragma unroll 4
  for (int it = 0; it < kChunk / 32; ++it) {
    const float4 a0 = *(const float4*)(pA + aoff);
    const float4 a1 = *(const float4*)(pA + aoff + 4);
    const float4 a2 = *(const float4*)(pA + aoff + 16);
    const float4 a3 = *(const float4*)(pA + aoff + 20);
    const float4 b0 = *(const float4*)(pB + boff);
    const float4 b1 = *(const float4*)(pB + boff + 4);
    const float4 b2 = *(const float4*)(pB + boff + 16);
    const float4 b3 = *(const float4*)(pB + boff + 20);
    v16h a, bb;
    a[0]=(_Float16)a0.x; a[1]=(_Float16)a0.y; a[2]=(_Float16)a0.z; a[3]=(_Float16)a0.w;
    a[4]=(_Float16)a1.x; a[5]=(_Float16)a1.y; a[6]=(_Float16)a1.z; a[7]=(_Float16)a1.w;
    a[8]=(_Float16)a2.x; a[9]=(_Float16)a2.y; a[10]=(_Float16)a2.z; a[11]=(_Float16)a2.w;
    a[12]=(_Float16)a3.x; a[13]=(_Float16)a3.y; a[14]=(_Float16)a3.z; a[15]=(_Float16)a3.w;
    bb[0]=(_Float16)b0.x; bb[1]=(_Float16)b0.y; bb[2]=(_Float16)b0.z; bb[3]=(_Float16)b0.w;
    bb[4]=(_Float16)b1.x; bb[5]=(_Float16)b1.y; bb[6]=(_Float16)b1.z; bb[7]=(_Float16)b1.w;
    bb[8]=(_Float16)b2.x; bb[9]=(_Float16)b2.y; bb[10]=(_Float16)b2.z; bb[11]=(_Float16)b2.w;
    bb[12]=(_Float16)b3.x; bb[13]=(_Float16)b3.y; bb[14]=(_Float16)b3.z; bb[15]=(_Float16)b3.w;
    acc = __builtin_amdgcn_wmma_f32_16x16x32_f16(
        /*neg_a=*/false, a, /*neg_b=*/false, bb,
        /*c_mod=*/(short)0, acc, /*reuse_a=*/false, /*reuse_b=*/false);
    pA += TOKSTRIDE;
    pB += TOKSTRIDE;
  }

  // C/D layout: VGPR v holds row (v + 8*half), col = lane&15
#pragma unroll
  for (int v = 0; v < 8; ++v)
    partial[w][(v + 8 * half) * 16 + mn] = acc[v];
  __syncthreads();

  const int m = tid >> 4, n = tid & 15;
  float sum = 0.0f;
#pragma unroll
  for (int ww = 0; ww < WAVES; ++ww) sum += partial[ww][tid];  // fixed order
  if (m < E_ && n < E_)
    costp[(b * KSPLIT + kb) * 36 + m * E_ + n] = -sum;
}

// ---------------------------------------------------------------------------
// Kernel 2: exact assignment over all 720 lexicographic permutations (Lehmer
// decode matches itertools.permutations order; ties -> smallest index = argmin).
// ---------------------------------------------------------------------------
__device__ __forceinline__ int lehmer_digit(int& rem, int t) {
  const int fact[5] = {120, 24, 6, 2, 1};
  if (t >= 5) return 0;
  int d = rem / fact[t];
  rem -= d * fact[t];
  return d;
}

__global__ __launch_bounds__(256) void perm_kernel(
    const float* __restrict__ costp, int* __restrict__ perms) {
  __shared__ float cost[36];
  __shared__ float bestVal[256];
  __shared__ int   bestIdx[256];
  const int b = blockIdx.x, tid = threadIdx.x;

  if (tid < 36) {
    float s = 0.0f;
    for (int kb = 0; kb < KSPLIT; ++kb) s += costp[(b * KSPLIT + kb) * 36 + tid];
    cost[tid] = s;
  }
  __syncthreads();

  float lbest = 3.402823466e38f;
  int   lidx  = 0;
  for (int p = tid; p < 720; p += 256) {
    int rem = p, avail = 0x543210;
    float tot = 0.0f;
#pragma unroll
    for (int t = 0; t < E_; ++t) {
      int d = lehmer_digit(rem, t);
      int o = (avail >> (4 * d)) & 0xF;
      int low  = avail & ((1 << (4 * d)) - 1);
      int high = (avail >> (4 * (d + 1))) << (4 * d);
      avail = low | high;
      tot += cost[t * E_ + o];
    }
    if (tot < lbest || (tot == lbest && p < lidx)) { lbest = tot; lidx = p; }
  }
  bestVal[tid] = lbest; bestIdx[tid] = lidx;
  __syncthreads();
  for (int off = 128; off > 0; off >>= 1) {
    if (tid < off) {
      float ov = bestVal[tid + off]; int oi = bestIdx[tid + off];
      if (ov < bestVal[tid] || (ov == bestVal[tid] && oi < bestIdx[tid])) {
        bestVal[tid] = ov; bestIdx[tid] = oi;
      }
    }
    __syncthreads();
  }
  if (tid == 0) {
    int rem = bestIdx[0], avail = 0x543210;
    for (int t = 0; t < E_; ++t) {
      int d = lehmer_digit(rem, t);
      int o = (avail >> (4 * d)) & 0xF;
      int low  = avail & ((1 << (4 * d)) - 1);
      int high = (avail >> (4 * (d + 1))) << (4 * d);
      avail = low | high;
      perms[b * E_ + t] = o;
    }
  }
}

// ---------------------------------------------------------------------------
// Kernel 3: masked BCE (torch -100 log clamp; x0.5 = mean over pair axis).
// 192 threads = one thread per intra-token offset r = e*32 + c*2 + i; each
// walks 64 tokens with a constant 192-float stride (no div/mod, coalesced).
// Uniform mask[tok] branch skips all traffic for masked-out tokens.
// ---------------------------------------------------------------------------
__global__ __launch_bounds__(BCE_THREADS) void bce_kernel(
    const float* __restrict__ outputs, const float* __restrict__ targets,
    const int* __restrict__ mask, const int* __restrict__ perms,
    float* __restrict__ partials) {
  __shared__ float red[BCE_THREADS];
  __shared__ int pm[E_];
  const int b = blockIdx.x >> 4;  // 16 blocks per batch (1024/64)
  if (threadIdx.x < E_) pm[threadIdx.x] = perms[b * E_ + threadIdx.x];
  __syncthreads();

  const int r   = threadIdx.x;    // 0..191
  const int e   = r >> 5;
  const int rem = r & 31;
  const int tokBase = blockIdx.x * TOK_PER_BLK;
  const float* tptr = targets + (size_t)tokBase * TOKSTRIDE + r;
  const float* optr = outputs + (size_t)tokBase * TOKSTRIDE + pm[e] * 32 + rem;

  float lsum = 0.0f;
  for (int it = 0; it < TOK_PER_BLK; ++it) {
    if (mask[tokBase + it]) {
      const float t = tptr[it * TOKSTRIDE];
      const float p = optr[it * TOKSTRIDE];
      const float lp  = fmaxf(logf(p),    -100.0f);
      const float l1p = fmaxf(log1pf(-p), -100.0f);
      lsum += -(t * lp + (1.0f - t) * l1p) * 0.5f;
    }
  }
  red[r] = lsum;
  __syncthreads();
  if (r < 64) red[r] += red[r + 128];   // 192 -> 128
  __syncthreads();
  for (int off = 64; off > 0; off >>= 1) {
    if (r < off) red[r] += red[r + off];
    __syncthreads();
  }
  if (r == 0) partials[blockIdx.x] = red[0];
}

// ---------------------------------------------------------------------------
// Kernel 4: deterministic final reduction: 512 partials + mask sum -> scalar.
// ---------------------------------------------------------------------------
__global__ __launch_bounds__(512) void finalize_kernel(
    const float* __restrict__ partials, const int* __restrict__ mask,
    float* __restrict__ outp) {
  __shared__ float red[512];
  __shared__ float redm[512];
  const int tid = threadIdx.x;
  float s = partials[tid];
  float ms = 0.0f;
  for (int t = tid; t < B_ * S_; t += 512) ms += (float)mask[t];
  red[tid] = s; redm[tid] = ms;
  __syncthreads();
  for (int off = 256; off > 0; off >>= 1) {
    if (tid < off) { red[tid] += red[tid + off]; redm[tid] += redm[tid + off]; }
    __syncthreads();
  }
  if (tid == 0) outp[0] = red[0] / redm[0];
}

extern "C" void kernel_launch(void* const* d_in, const int* in_sizes, int n_in,
                              void* d_out, int out_size, void* d_ws, size_t ws_size,
                              hipStream_t stream) {
  const float* outputs = (const float*)d_in[0];   // [B,S,E,C,2] f32
  const float* targets = (const float*)d_in[1];   // [B,S,E,C,2] f32
  const int*   mask    = (const int*)d_in[2];     // [B,S] i32

  char* ws = (char*)d_ws;
  float* costp    = (float*)ws;                                     // 32*8*36 f32
  int*   perms    = (int*)(ws + (size_t)B_ * KSPLIT * 36 * 4);      // 32*6 i32
  float* partials = (float*)(ws + (size_t)B_ * KSPLIT * 36 * 4 + (size_t)B_ * E_ * 4);

  dim3 g1(B_, KSPLIT);
  cost_wmma_kernel<<<g1, 256, 0, stream>>>(outputs, targets, costp);
  perm_kernel<<<B_, 256, 0, stream>>>(costp, perms);
  bce_kernel<<<(B_ * S_) / TOK_PER_BLK, BCE_THREADS, 0, stream>>>(outputs, targets, mask, perms, partials);
  finalize_kernel<<<1, 512, 0, stream>>>(partials, mask, (float*)d_out);
}
